// MultiHeadSelfAttention_35613868818482
// MI455X (gfx1250) — compile-verified, matrix operand order checked by reading, not run on hardware
//
#include <hip/hip_runtime.h>
#include <hip/hip_bf16.h>
#include <stdint.h>

// ---------------------------------------------------------------------------
// MI455X (gfx1250) fused MHSA: bf16 WMMA GEMMs + causal flash attention.
// wave32; v_wmma_f32_16x16x32_bf16; TDM (tensor_load_to_lds) tile staging,
// double-buffered, tracked with TENSORcnt.
// ---------------------------------------------------------------------------

typedef __attribute__((ext_vector_type(16))) __bf16 v16bf;
typedef __attribute__((ext_vector_type(8)))  __bf16 v8bf;
typedef __attribute__((ext_vector_type(4)))  __bf16 v4bf;
typedef __attribute__((ext_vector_type(8)))  float  v8f;
typedef __attribute__((ext_vector_type(4)))  float  v4f;
typedef __attribute__((ext_vector_type(4)))  unsigned int v4u;
typedef __attribute__((ext_vector_type(8)))  int    v8i;
typedef __attribute__((ext_vector_type(4)))  int    v4i;

#define D_MODEL 1024
#define SEQ     2048
#define BATCH   4
#define HEADS   16
#define DKH     64
#define MTOT    (BATCH * SEQ)     // 8192

__device__ __forceinline__ v8f wmma_bf16(v16bf a, v16bf b, v8f c) {
  // D = A(16x32 bf16) * B(32x16 bf16) + C(16x16 f32)
  return __builtin_amdgcn_wmma_f32_16x16x32_bf16(
      /*neg_a=*/false, a, /*neg_b=*/false, b,
      /*c_mod=*/(short)0, c, /*reuse_a=*/false, /*reuse_b=*/false);
}

// Assemble a 16x32 A/B fragment per the ISA layout:
// lane holds 8 contiguous halfs at k-offset (lane/16)*8 and another 8 at +16.
__device__ __forceinline__ v16bf ldfrag(const __bf16* p) {
  v8bf lo = *(const v8bf*)(p);
  v8bf hi = *(const v8bf*)(p + 16);
  return __builtin_shufflevector(lo, hi,
      0, 1, 2, 3, 4, 5, 6, 7, 8, 9, 10, 11, 12, 13, 14, 15);
}

__device__ __forceinline__ void lds_fence_wave() {
  asm volatile("s_wait_dscnt 0" ::: "memory");
  __builtin_amdgcn_wave_barrier();
}

// ---------------------------------------------------------------------------
// TDM: one instruction DMAs a 2-D tile (rows x 32 bf16) from global into LDS,
// re-padding each 64B row to a 96B LDS pitch (pad_interval=16dw, pad=8dw) so
// the WMMA fragment loaders keep their bank-friendly [rows][48] layout.
// ---------------------------------------------------------------------------
__device__ __forceinline__ void tdm_load_tile_2d(
    const void* gaddr, unsigned lds_addr,
    unsigned tensor_d0, unsigned tensor_d1,
    unsigned tile_d0, unsigned tile_d1, unsigned stride0) {
  uint64_t ga = (uint64_t)gaddr;
  v4u g0;
  g0[0] = 1u;                                   // count=1, user descriptor
  g0[1] = lds_addr;                             // LDS byte address
  g0[2] = (unsigned)ga;                         // global_addr[31:0]
  g0[3] = ((unsigned)(ga >> 32) & 0x01FFFFFFu)  // global_addr[56:32]
          | (2u << 30);                         // type=2 ("image")
  v8i g1;
  g1[0] = (int)((1u << 16)        // data_size = 1 -> 2 bytes/elem
              | (1u << 20)        // pad_enable
              | (3u << 22)        // pad_interval: 16 DWORDs (64B) per burst
              | (7u << 25));      // pad_amount:   8 DWORDs (32B) pad
  g1[1] = (int)((tensor_d0 & 0xFFFFu) << 16);                    // td0[15:0]
  g1[2] = (int)((tensor_d0 >> 16) | ((tensor_d1 & 0xFFFFu) << 16));
  g1[3] = (int)((tensor_d1 >> 16) | (tile_d0 << 16));            // tile_dim0
  g1[4] = (int)(tile_d1 & 0xFFFFu);                              // tile_dim1
  g1[5] = (int)stride0;                                          // dim0 stride
  g1[6] = 0;
  g1[7] = 0;
  v4i z4 = {0, 0, 0, 0};
#if __clang_major__ >= 23
  v8i z8 = {0, 0, 0, 0, 0, 0, 0, 0};
  __builtin_amdgcn_tensor_load_to_lds(g0, g1, z4, z4, z8, 0);
#else
  __builtin_amdgcn_tensor_load_to_lds(g0, g1, z4, z4, 0);
#endif
}

// ---------------------------------------------------------------------------
// fp32 -> bf16 convert (vectorized: b128 load / b64 store)
// ---------------------------------------------------------------------------
__global__ void f32_to_bf16_kernel(const float* __restrict__ in,
                                   __bf16* __restrict__ out, int n) {
  int n4 = n >> 2;
  for (int i = blockIdx.x * 256 + threadIdx.x; i < n4; i += gridDim.x * 256) {
    v4f v = ((const v4f*)in)[i];
    v4bf o;
    o.x = (__bf16)v.x; o.y = (__bf16)v.y; o.z = (__bf16)v.z; o.w = (__bf16)v.w;
    ((v4bf*)out)[i] = o;
  }
}

// ---------------------------------------------------------------------------
// C = A[8192,1024] * W^T, W[1024,1024] row-major (B[k][n] = W[n][k]).
// Block tile 128x256, 8 waves in a 2x4 grid, each wave 64x64 (4x4 WMMA accs).
// Double-buffered LDS filled by the Tensor Data Mover (2 TDM ops per tile,
// issued by wave 0), overlapped with WMMA.
// mode 0: fp32 out [M,1024]
// mode 1: bf16 out [b,h,s,dk]  (Q with scale=0.125, K with scale=1)
// mode 2: bf16 out [b,h,dk,s]  (V transposed for the PV matmul)
// ---------------------------------------------------------------------------
__global__ __launch_bounds__(256) void gemm_x_wT_kernel(
    const __bf16* __restrict__ A, const __bf16* __restrict__ W,
    void* __restrict__ Cout, int mode, float scale) {
  __shared__ __attribute__((aligned(16))) __bf16 As[2][128][48];  // 24 KB
  __shared__ __attribute__((aligned(16))) __bf16 Bs[2][256][48];  // 48 KB

  const int tid  = threadIdx.x;
  const int lane = tid & 31, wid = tid >> 5;
  const int wm = wid >> 2, wn = wid & 3;           // 2 x 4 wave grid
  const int hl = lane >> 4, lm = lane & 15;
  const int m0 = blockIdx.y * 128, n0 = blockIdx.x * 256;

  // Wave 0 drives the TDM: one descriptor per tile (A: 128x32, B: 256x32).
  auto issue_tile = [&](int kbi, int bufi) {
    if (tid < 32) {
      tdm_load_tile_2d(&A[(size_t)m0 * D_MODEL + kbi * 32],
                       (unsigned)(uintptr_t)&As[bufi][0][0],
                       D_MODEL, MTOT,    32, 128, D_MODEL);
      tdm_load_tile_2d(&W[(size_t)n0 * D_MODEL + kbi * 32],
                       (unsigned)(uintptr_t)&Bs[bufi][0][0],
                       D_MODEL, D_MODEL, 32, 256, D_MODEL);
    }
  };

  v8f acc[4][4] = {};
  constexpr int KSTEPS = D_MODEL / 32;

  issue_tile(0, 0);
  for (int kb = 0; kb < KSTEPS; ++kb) {
    const int cur = kb & 1;
    if (kb + 1 < KSTEPS) {
      issue_tile(kb + 1, cur ^ 1);     // prefetch next tile into other buffer
      if (tid < 32) __builtin_amdgcn_s_wait_tensorcnt(2);  // tile kb done
    } else {
      if (tid < 32) __builtin_amdgcn_s_wait_tensorcnt(0);
    }
    __syncthreads();                   // TDM data visible to all waves

    v16bf af[4], bfr[4];
#pragma unroll
    for (int mt = 0; mt < 4; ++mt)
      af[mt] = ldfrag(&As[cur][wm * 64 + mt * 16 + lm][hl * 8]);
#pragma unroll
    for (int nt = 0; nt < 4; ++nt)
      bfr[nt] = ldfrag(&Bs[cur][wn * 64 + nt * 16 + lm][hl * 8]);
#pragma unroll
    for (int mt = 0; mt < 4; ++mt)
#pragma unroll
      for (int nt = 0; nt < 4; ++nt)
        acc[mt][nt] = wmma_bf16(af[mt], bfr[nt], acc[mt][nt]);
    __syncthreads();                   // reads done before buffer is refilled
  }

  // Epilogue. C layout: reg r, lane l -> row r + 8*(l/16), col l%16.
#pragma unroll
  for (int mt = 0; mt < 4; ++mt)
#pragma unroll
    for (int nt = 0; nt < 4; ++nt)
#pragma unroll
      for (int r = 0; r < 8; ++r) {
        float v = acc[mt][nt][r] * scale;
        int gm = m0 + wm * 64 + mt * 16 + r + hl * 8;
        int gn = n0 + wn * 64 + nt * 16 + lm;
        if (mode == 0) {
          ((float*)Cout)[(size_t)gm * D_MODEL + gn] = v;
        } else {
          int b = gm >> 11, s = gm & (SEQ - 1);
          int h = gn >> 6,  dh = gn & (DKH - 1);
          __bf16 bv = (__bf16)v;
          if (mode == 1)
            ((__bf16*)Cout)[(((size_t)b * HEADS + h) * SEQ + s) * DKH + dh] = bv;
          else
            ((__bf16*)Cout)[(((size_t)b * HEADS + h) * DKH + dh) * SEQ + s] = bv;
        }
      }
}

// ---------------------------------------------------------------------------
// Flash attention (causal). grid = (S/128, B*H); 8 waves, wave w owns
// 16 query rows q0 = blockIdx.x*128 + w*16. Q pre-scaled by 1/sqrt(dk).
// ---------------------------------------------------------------------------
__global__ __launch_bounds__(256) void attn_kernel(
    const __bf16* __restrict__ Q,   // [bh][s][64]
    const __bf16* __restrict__ Kg,  // [bh][s][64]
    const __bf16* __restrict__ Vt,  // [bh][64][s]
    __bf16* __restrict__ Oa) {      // [b][s][1024]
  __shared__ __attribute__((aligned(16))) __bf16 Ps[8][16][40];  // per-wave P

  const int tid = threadIdx.x, lane = tid & 31, w = tid >> 5;
  const int hl = lane >> 4, lm = lane & 15;
  const int bh = blockIdx.y;
  const int bidx = bh >> 4, h = bh & 15;
  const int q0 = blockIdx.x * 128 + w * 16;

  const __bf16* Qb = Q  + (size_t)bh * SEQ * DKH;
  const __bf16* Kb = Kg + (size_t)bh * SEQ * DKH;
  const __bf16* Vb = Vt + (size_t)bh * DKH * SEQ;

  // Q A-fragments (16 rows x dk=64 -> two 16x32 frags), kept in VGPRs.
  v16bf qa0, qa1;
  {
    const __bf16* qp = &Qb[(size_t)(q0 + lm) * DKH + hl * 8];
    qa0 = ldfrag(qp);
    qa1 = ldfrag(qp + 32);
  }

  float mrow[8], lrow[8];
  v8f oacc[4] = {};
#pragma unroll
  for (int r = 0; r < 8; ++r) { mrow[r] = -1e30f; lrow[r] = 0.f; }

  const int nk = q0 + 16;            // causal: need keys [0, nk)
  const int nb = (nk + 31) >> 5;     // 32-key blocks

  for (int kb = 0; kb < nb; ++kb) {
    const int k0 = kb * 32;

    // ---- S = Q * K^T for two 16-key tiles --------------------------------
    v8f st[2];
#pragma unroll
    for (int t = 0; t < 2; ++t) {
      int key = k0 + t * 16 + lm;
      const __bf16* kp = &Kb[(size_t)key * DKH + hl * 8];
      v16bf kf0 = ldfrag(kp);
      v16bf kf1 = ldfrag(kp + 32);
      v8f s = {};
      s = wmma_bf16(qa0, kf0, s);
      s = wmma_bf16(qa1, kf1, s);
      st[t] = s;
    }

    // ---- causal mask ------------------------------------------------------
#pragma unroll
    for (int r = 0; r < 8; ++r) {
      int q = q0 + r + hl * 8;
      if (k0 + lm > q)      st[0][r] = -1e30f;
      if (k0 + 16 + lm > q) st[1][r] = -1e30f;
    }

    // ---- online softmax (16-lane segmented reductions) --------------------
#pragma unroll
    for (int r = 0; r < 8; ++r) {
      float v = fmaxf(st[0][r], st[1][r]);
      for (int off = 8; off >= 1; off >>= 1)
        v = fmaxf(v, __shfl_xor(v, off, 32));
      float mn = fmaxf(mrow[r], v);
      float es = __expf(mrow[r] - mn);
      mrow[r] = mn;
      float p0 = __expf(st[0][r] - mn);
      float p1 = __expf(st[1][r] - mn);
      st[0][r] = p0; st[1][r] = p1;
      float rs = p0 + p1;
      for (int off = 8; off >= 1; off >>= 1)
        rs += __shfl_xor(rs, off, 32);
      lrow[r] = lrow[r] * es + rs;
#pragma unroll
      for (int nt = 0; nt < 4; ++nt) oacc[nt][r] *= es;
    }

    // ---- transpose P (C-layout -> A-layout) via wave-private LDS ----------
#pragma unroll
    for (int r = 0; r < 8; ++r) {
      int row = r + hl * 8;
      Ps[w][row][lm]      = (__bf16)st[0][r];
      Ps[w][row][16 + lm] = (__bf16)st[1][r];
    }
    lds_fence_wave();                      // DS in-order within wave
    v16bf pa = ldfrag(&Ps[w][lm][hl * 8]); // A-frag: 16 q-rows x 32 keys

    // ---- O += P * V (V transposed: rows = d, contiguous in keys) ----------
#pragma unroll
    for (int nt = 0; nt < 4; ++nt) {
      int d = nt * 16 + lm;
      v16bf vf = ldfrag(&Vb[(size_t)d * SEQ + k0 + hl * 8]);
      oacc[nt] = wmma_bf16(pa, vf, oacc[nt]);
    }
  }

  // ---- normalize + store [b, s, h*64+d] as bf16 ---------------------------
#pragma unroll
  for (int r = 0; r < 8; ++r) {
    float inv = 1.f / lrow[r];
    int s = q0 + r + hl * 8;
    size_t base = ((size_t)bidx * SEQ + s) * D_MODEL + h * DKH;
#pragma unroll
    for (int nt = 0; nt < 4; ++nt)
      Oa[base + nt * 16 + lm] = (__bf16)(oacc[nt][r] * inv);
  }
}

// ---------------------------------------------------------------------------
// Host launch
// ---------------------------------------------------------------------------
extern "C" void kernel_launch(void* const* d_in, const int* in_sizes, int n_in,
                              void* d_out, int out_size, void* d_ws, size_t ws_size,
                              hipStream_t stream) {
  const float* x  = (const float*)d_in[0];
  const float* Wq = (const float*)d_in[1];
  const float* Wk = (const float*)d_in[2];
  const float* Wv = (const float*)d_in[3];
  const float* Wo = (const float*)d_in[4];
  float* out = (float*)d_out;

  char* ws = (char*)d_ws;
  const size_t XB  = (size_t)MTOT * D_MODEL * 2;       // 16 MB (also reused as Oa)
  const size_t WB  = (size_t)D_MODEL * D_MODEL * 2;    // 2 MB
  const size_t QKV = (size_t)MTOT * D_MODEL * 2;       // 16 MB each

  __bf16* xb  = (__bf16*)(ws);                         // X bf16; later aliased by Oa
  __bf16* wqb = (__bf16*)(ws + XB);
  __bf16* wkb = (__bf16*)(ws + XB + WB);
  __bf16* wvb = (__bf16*)(ws + XB + 2 * WB);
  __bf16* wob = (__bf16*)(ws + XB + 3 * WB);
  __bf16* Qb  = (__bf16*)(ws + XB + 4 * WB);
  __bf16* Kb  = (__bf16*)(ws + XB + 4 * WB + QKV);
  __bf16* Vtb = (__bf16*)(ws + XB + 4 * WB + 2 * QKV);
  __bf16* Oa  = xb;  // X no longer needed once Q/K/V are built

  // 1) fp32 -> bf16
  f32_to_bf16_kernel<<<2048, 256, 0, stream>>>(x,  xb,  MTOT * D_MODEL);
  f32_to_bf16_kernel<<<1024, 256, 0, stream>>>(Wq, wqb, D_MODEL * D_MODEL);
  f32_to_bf16_kernel<<<1024, 256, 0, stream>>>(Wk, wkb, D_MODEL * D_MODEL);
  f32_to_bf16_kernel<<<1024, 256, 0, stream>>>(Wv, wvb, D_MODEL * D_MODEL);
  f32_to_bf16_kernel<<<1024, 256, 0, stream>>>(Wo, wob, D_MODEL * D_MODEL);

  // 2) Q/K/V projections (Q pre-scaled by 1/sqrt(64))
  dim3 ggrid(D_MODEL / 256, MTOT / 128);   // (4, 64)
  gemm_x_wT_kernel<<<ggrid, 256, 0, stream>>>(xb, wqb, Qb,  1, 0.125f);
  gemm_x_wT_kernel<<<ggrid, 256, 0, stream>>>(xb, wkb, Kb,  1, 1.0f);
  gemm_x_wT_kernel<<<ggrid, 256, 0, stream>>>(xb, wvb, Vtb, 2, 1.0f);

  // 3) causal flash attention
  attn_kernel<<<dim3(SEQ / 128, BATCH * HEADS), 256, 0, stream>>>(Qb, Kb, Vtb, Oa);

  // 4) output projection -> fp32 d_out
  gemm_x_wT_kernel<<<ggrid, 256, 0, stream>>>(Oa, wob, (void*)out, 0, 1.0f);
}
